// trajectory2seq_59115929862754
// MI455X (gfx1250) — compile-verified, matrix operand order checked by reading, not run on hardware
//
#include <hip/hip_runtime.h>
#include <hip/hip_bf16.h>
#include <math.h>

// ---------------------------------------------------------------------------
// Model dims
// ---------------------------------------------------------------------------
#define BB    128          // batch
#define TT    512          // encoder time steps
#define HH    256          // hidden
#define GG    768          // 3*H gate width
#define VV    29           // vocab
#define STEPS 63           // decoder steps (64-1)

typedef __attribute__((ext_vector_type(16))) _Float16 v16h;
typedef __attribute__((ext_vector_type(8)))  float    v8f;

// ---------------------------------------------------------------------------
// WMMA helpers (CDNA5 wave32, v_wmma_f32_16x16x32_f16)
// A: 16x32 f16 (one output-row tile), B: 32x16 f16 (one weight-row tile,
// weights stored (N,K) row-major so column n of B == weight row n, K-contig).
// ---------------------------------------------------------------------------
static __device__ __forceinline__ v8f wmma32(v16h a, v16h b, v8f c) {
  return __builtin_amdgcn_wmma_f32_16x16x32_f16(
      /*neg_a=*/false, a, /*neg_b=*/false, b,
      /*c_mod=*/(short)0, c, /*reuse_a=*/false, /*reuse_b=*/false);
}

// Load a 16x32 f16 fragment: lane L holds 16 contiguous halves of
// row (L&15), K-half selected by lane[4].  ld = row stride in elements.
static __device__ __forceinline__ v16h ldfrag(const _Float16* p, int ld) {
  const int lane = threadIdx.x & 31;
  return *(const v16h*)(p + (size_t)(lane & 15) * ld + ((lane >> 4) << 4));
}

// Store a 16x16 f32 C/D tile (ISA layout: VGPR r -> M = r + 8*(lane>=16),
// N = lane&15) to row-major f32 buffer.
static __device__ __forceinline__ void tstore(float* C, int ld, v8f acc) {
  const int lane = threadIdx.x & 31;
  const int col  = lane & 15;
  const int row0 = (lane >> 4) << 3;
#pragma unroll
  for (int r = 0; r < 8; ++r) C[(size_t)(row0 + r) * ld + col] = acc[r];
}

static __device__ __forceinline__ float sigmoidf_(float x) {
  return 1.0f / (1.0f + __expf(-x));
}

// ---------------------------------------------------------------------------
// fp32 -> fp16 conversion
// ---------------------------------------------------------------------------
__global__ void cvt_f32_to_f16(const float* __restrict__ s,
                               _Float16* __restrict__ d, int n) {
  int i = blockIdx.x * blockDim.x + threadIdx.x;
  if (i < n) d[i] = (_Float16)s[i];
}

// ---------------------------------------------------------------------------
// Generic tiled WMMA GEMM:  C[M,N] = A[M,K] @ W[N,K]^T + bias[N]   (f32 out)
// block = 128 threads (4 waves), grid = (M/16, N/64); wave j -> n-tile.
// ---------------------------------------------------------------------------
__global__ void __launch_bounds__(128)
wmma_gemm_abt(const _Float16* __restrict__ A, const _Float16* __restrict__ W,
              const float* __restrict__ bias, float* __restrict__ C,
              int M, int N, int K) {
  const int m0 = blockIdx.x << 4;
  const int wave = threadIdx.x >> 5;
  const int n0 = ((blockIdx.y << 2) + wave) << 4;
  v8f acc = {};
  for (int k = 0; k < K; k += 32) {
    __builtin_prefetch(W + (size_t)n0 * K + k + 512, 0, 1);
    v16h a = ldfrag(A + (size_t)m0 * K + k, K);
    v16h b = ldfrag(W + (size_t)n0 * K + k, K);
    acc = wmma32(a, b, acc);
  }
  const int lane = threadIdx.x & 31;
  const int col  = lane & 15;
  const int row0 = (lane >> 4) << 3;
  const float bv = bias[n0 + col];
#pragma unroll
  for (int r = 0; r < 8; ++r)
    C[(size_t)(m0 + row0 + r) * N + n0 + col] = acc[r] + bv;
}

// ---------------------------------------------------------------------------
// Encoder layer 0 GRU scan.  grid = 2 (dir), block = 1024 (32 waves).
// Per step: gh = h @ Whh^T via WMMA (h in LDS as f16), then gate phase does
// the K=2 input projection scalar and the GRU update, writing y0 (f16 concat).
// ---------------------------------------------------------------------------
__global__ void __launch_bounds__(1024)
enc_scan_l0(const float* __restrict__ x, const float* __restrict__ Wih0,
            const float* __restrict__ bih0, const _Float16* __restrict__ Whh0h,
            const float* __restrict__ bhh0, _Float16* __restrict__ y0,
            float* __restrict__ hT, float* __restrict__ ghws) {
  const int dir = blockIdx.x;
  const float*    Wih = Wih0 + (size_t)dir * GG * 2;
  const float*    bih = bih0 + (size_t)dir * GG;
  const _Float16* Whh = Whh0h + (size_t)dir * GG * HH;
  const float*    bhh = bhh0 + (size_t)dir * GG;
  float*          gh  = ghws + (size_t)dir * BB * GG;
  float*          hTo = hT + (size_t)dir * BB * HH;

  __shared__ _Float16 hs[BB * HH];  // 64 KB hidden state
  const int tid = threadIdx.x;
  for (int i = tid; i < BB * HH; i += 1024) hs[i] = (_Float16)0.0f;
  __syncthreads();

  const int wave = tid >> 5;
  const int m0   = (wave >> 2) << 4;  // 8 M-tiles
  const int ng   = wave & 3;          // 4 N-groups x 12 tiles = 48 N-tiles

  for (int s = 0; s < TT; ++s) {
    const int t = dir ? (TT - 1 - s) : s;
    // phase 1: gh = h @ Whh^T   (128 x 768 x 256)
    for (int j = 0; j < 12; ++j) {
      const int n0 = ((ng * 12) + j) << 4;
      v8f acc = {};
#pragma unroll
      for (int kk = 0; kk < HH; kk += 32) {
        v16h a = ldfrag(hs + (size_t)m0 * HH + kk, HH);   // LDS
        v16h b = ldfrag(Whh + (size_t)n0 * HH + kk, HH);  // global (L2-resident)
        acc = wmma32(a, b, acc);
      }
      tstore(gh + (size_t)m0 * GG + n0, GG, acc);
    }
    __threadfence_block();
    __syncthreads();
    // phase 2: gates (input projection K=2 done scalar)
    for (int e = tid; e < BB * HH; e += 1024) {
      const int b = e >> 8, c = e & 255;
      const float x0 = x[((size_t)b * TT + t) * 2 + 0];
      const float x1 = x[((size_t)b * TT + t) * 2 + 1];
      const float ir  = x0 * Wih[2 * c]           + x1 * Wih[2 * c + 1]           + bih[c];
      const float iz  = x0 * Wih[2 * (256 + c)]   + x1 * Wih[2 * (256 + c) + 1]   + bih[256 + c];
      const float inn = x0 * Wih[2 * (512 + c)]   + x1 * Wih[2 * (512 + c) + 1]   + bih[512 + c];
      const float hr = gh[(size_t)b * GG + c]       + bhh[c];
      const float hz = gh[(size_t)b * GG + 256 + c] + bhh[256 + c];
      const float hn = gh[(size_t)b * GG + 512 + c] + bhh[512 + c];
      const float r = sigmoidf_(ir + hr);
      const float z = sigmoidf_(iz + hz);
      const float nn = tanhf(inn + r * hn);
      const float hold = (float)hs[e];
      const float hnew = (1.0f - z) * nn + z * hold;
      hs[e] = (_Float16)hnew;
      y0[((size_t)b * TT + t) * 512 + dir * 256 + c] = (_Float16)hnew;
      if (s == TT - 1) hTo[e] = hnew;
    }
    __threadfence_block();
    __syncthreads();
  }
}

// ---------------------------------------------------------------------------
// Encoder layer 1 GRU scan.  Input x_t = y0[:, t, :] (512-wide, f16).
// Per step: gi = x_t @ Wih1^T (K=512) and gh = h @ Whh1^T (K=256), both WMMA
// (kept separate: n-gate needs inn and hn independently).
// ---------------------------------------------------------------------------
__global__ void __launch_bounds__(1024)
enc_scan_l1(const _Float16* __restrict__ y0, const _Float16* __restrict__ Wih1h,
            const float* __restrict__ bih1, const _Float16* __restrict__ Whh1h,
            const float* __restrict__ bhh1, _Float16* __restrict__ y1,
            float* __restrict__ hT, float* __restrict__ giws,
            float* __restrict__ ghws) {
  const int dir = blockIdx.x;
  const _Float16* Wih = Wih1h + (size_t)dir * GG * 512;
  const float*    bih = bih1 + (size_t)dir * GG;
  const _Float16* Whh = Whh1h + (size_t)dir * GG * HH;
  const float*    bhh = bhh1 + (size_t)dir * GG;
  float*          gi  = giws + (size_t)dir * BB * GG;
  float*          gh  = ghws + (size_t)dir * BB * GG;
  float*          hTo = hT + (size_t)dir * BB * HH;

  __shared__ _Float16 hs[BB * HH];
  const int tid = threadIdx.x;
  for (int i = tid; i < BB * HH; i += 1024) hs[i] = (_Float16)0.0f;
  __syncthreads();

  const int wave = tid >> 5;
  const int m0   = (wave >> 2) << 4;
  const int ng   = wave & 3;

  for (int s = 0; s < TT; ++s) {
    const int t = dir ? (TT - 1 - s) : s;
    for (int j = 0; j < 12; ++j) {
      const int n0 = ((ng * 12) + j) << 4;
      // gh = h @ Whh^T  (K = 256)
      v8f ah = {};
#pragma unroll
      for (int kk = 0; kk < HH; kk += 32) {
        v16h a = ldfrag(hs + (size_t)m0 * HH + kk, HH);
        v16h b = ldfrag(Whh + (size_t)n0 * HH + kk, HH);
        ah = wmma32(a, b, ah);
      }
      tstore(gh + (size_t)m0 * GG + n0, GG, ah);
      // gi = x_t @ Wih^T  (K = 512); A rows are batch rows of y0 at time t
      v8f ai = {};
#pragma unroll
      for (int kk = 0; kk < 512; kk += 32) {
        v16h a = ldfrag(y0 + ((size_t)m0 * TT + t) * 512 + kk, TT * 512);
        v16h b = ldfrag(Wih + (size_t)n0 * 512 + kk, 512);
        ai = wmma32(a, b, ai);
      }
      tstore(gi + (size_t)m0 * GG + n0, GG, ai);
    }
    __threadfence_block();
    __syncthreads();
    for (int e = tid; e < BB * HH; e += 1024) {
      const int b = e >> 8, c = e & 255;
      const float ir  = gi[(size_t)b * GG + c]       + bih[c];
      const float iz  = gi[(size_t)b * GG + 256 + c] + bih[256 + c];
      const float inn = gi[(size_t)b * GG + 512 + c] + bih[512 + c];
      const float hr = gh[(size_t)b * GG + c]       + bhh[c];
      const float hz = gh[(size_t)b * GG + 256 + c] + bhh[256 + c];
      const float hn = gh[(size_t)b * GG + 512 + c] + bhh[512 + c];
      const float r = sigmoidf_(ir + hr);
      const float z = sigmoidf_(iz + hz);
      const float nn = tanhf(inn + r * hn);
      const float hold = (float)hs[e];
      const float hnew = (1.0f - z) * nn + z * hold;
      hs[e] = (_Float16)hnew;
      y1[((size_t)b * TT + t) * 512 + dir * 256 + c] = (_Float16)hnew;
      if (s == TT - 1) hTo[e] = hnew;
    }
    __threadfence_block();
    __syncthreads();
  }
}

// ---------------------------------------------------------------------------
// Persistent decoder: 63 steps, single workgroup (256 threads = 8 waves),
// greedy argmax feedback.  GEMMs via WMMA; attention/softmax/logits via VALU.
// ---------------------------------------------------------------------------
__global__ void __launch_bounds__(256)
decoder_kernel(const int* __restrict__ tgt, const float* __restrict__ emb,
               const float* __restrict__ dWih0, const _Float16* __restrict__ dWih1h,
               const _Float16* __restrict__ dWhhh, const float* __restrict__ dbih,
               const float* __restrict__ dbhh, const _Float16* __restrict__ qWh,
               const float* __restrict__ qb, const float* __restrict__ outW,
               const float* __restrict__ outb, const float* __restrict__ hT0,
               const float* __restrict__ hT1, const float* __restrict__ enc,
               _Float16* __restrict__ h0h, _Float16* __restrict__ h1h,
               float* __restrict__ h0f, float* __restrict__ h1f,
               float* __restrict__ gi_s, float* __restrict__ gh_s,
               float* __restrict__ q_s, float* __restrict__ ctxb,
               float* __restrict__ sc, float* __restrict__ out) {
  __shared__ int tok[BB];
  const int tid  = threadIdx.x;
  const int wave = tid >> 5;
  const int lane = tid & 31;

  // init: h0 = hf0 + hb0 ; h1 = hf1 + hb1 ; tok = target_seq[:,0]
  for (int i = tid; i < BB * HH; i += 256) {
    const float a = hT0[i] + hT0[BB * HH + i];
    const float b = hT1[i] + hT1[BB * HH + i];
    h0f[i] = a; h0h[i] = (_Float16)a;
    h1f[i] = b; h1h[i] = (_Float16)b;
  }
  for (int i = tid; i < BB; i += 256) tok[i] = tgt[(size_t)i * 64];
  __threadfence_block();
  __syncthreads();

  for (int step = 0; step < STEPS; ++step) {
    // ---- GRU cell 0: gh0 = h0 @ dWhh[0]^T --------------------------------
    for (int j = 0; j < 48; ++j) {
      const int m0 = wave << 4, n0 = j << 4;
      v8f acc = {};
#pragma unroll
      for (int kk = 0; kk < HH; kk += 32)
        acc = wmma32(ldfrag(h0h + (size_t)m0 * HH + kk, HH),
                     ldfrag(dWhhh + (size_t)n0 * HH + kk, HH), acc);
      tstore(gh_s + (size_t)m0 * GG + n0, GG, acc);
    }
    __threadfence_block();
    __syncthreads();
    // gates 0 (input projection K=8 scalar: e = emb[tok])
    for (int e = tid; e < BB * HH; e += 256) {
      const int b = e >> 8, c = e & 255;
      const float* ev = emb + (size_t)tok[b] * 8;
      float ir = dbih[c], iz = dbih[256 + c], inn = dbih[512 + c];
#pragma unroll
      for (int k = 0; k < 8; ++k) {
        ir  += ev[k] * dWih0[(size_t)c * 8 + k];
        iz  += ev[k] * dWih0[(size_t)(256 + c) * 8 + k];
        inn += ev[k] * dWih0[(size_t)(512 + c) * 8 + k];
      }
      const float hr = gh_s[(size_t)b * GG + c]       + dbhh[c];
      const float hz = gh_s[(size_t)b * GG + 256 + c] + dbhh[256 + c];
      const float hn = gh_s[(size_t)b * GG + 512 + c] + dbhh[512 + c];
      const float r = sigmoidf_(ir + hr);
      const float z = sigmoidf_(iz + hz);
      const float nn = tanhf(inn + r * hn);
      const float hnew = (1.0f - z) * nn + z * h0f[e];
      h0f[e] = hnew; h0h[e] = (_Float16)hnew;
    }
    __threadfence_block();
    __syncthreads();
    // ---- GRU cell 1: gi1 = h0 @ dWih1^T ; gh1 = h1 @ dWhh[1]^T -----------
    for (int j = 0; j < 48; ++j) {
      const int m0 = wave << 4, n0 = j << 4;
      v8f ai = {}, ah = {};
#pragma unroll
      for (int kk = 0; kk < HH; kk += 32) {
        ai = wmma32(ldfrag(h0h + (size_t)m0 * HH + kk, HH),
                    ldfrag(dWih1h + (size_t)n0 * HH + kk, HH), ai);
        ah = wmma32(ldfrag(h1h + (size_t)m0 * HH + kk, HH),
                    ldfrag(dWhhh + (size_t)GG * HH + (size_t)n0 * HH + kk, HH), ah);
      }
      tstore(gi_s + (size_t)m0 * GG + n0, GG, ai);
      tstore(gh_s + (size_t)m0 * GG + n0, GG, ah);
    }
    __threadfence_block();
    __syncthreads();
    // gates 1
    for (int e = tid; e < BB * HH; e += 256) {
      const int b = e >> 8, c = e & 255;
      const float ir  = gi_s[(size_t)b * GG + c]       + dbih[GG + c];
      const float iz  = gi_s[(size_t)b * GG + 256 + c] + dbih[GG + 256 + c];
      const float inn = gi_s[(size_t)b * GG + 512 + c] + dbih[GG + 512 + c];
      const float hr = gh_s[(size_t)b * GG + c]       + dbhh[GG + c];
      const float hz = gh_s[(size_t)b * GG + 256 + c] + dbhh[GG + 256 + c];
      const float hn = gh_s[(size_t)b * GG + 512 + c] + dbhh[GG + 512 + c];
      const float r = sigmoidf_(ir + hr);
      const float z = sigmoidf_(iz + hz);
      const float nn = tanhf(inn + r * hn);
      const float hnew = (1.0f - z) * nn + z * h1f[e];
      h1f[e] = hnew; h1h[e] = (_Float16)hnew;
    }
    __threadfence_block();
    __syncthreads();
    // ---- q = h1 @ q_W^T + q_b  (128x256x256) -----------------------------
    for (int j = 0; j < 16; ++j) {
      const int m0 = wave << 4, n0 = j << 4;
      v8f acc = {};
#pragma unroll
      for (int kk = 0; kk < HH; kk += 32)
        acc = wmma32(ldfrag(h1h + (size_t)m0 * HH + kk, HH),
                     ldfrag(qWh + (size_t)n0 * HH + kk, HH), acc);
      const int col = lane & 15, row0 = (lane >> 4) << 3;
      const float bv = qb[n0 + col];
#pragma unroll
      for (int r = 0; r < 8; ++r)
        q_s[(size_t)(m0 + row0 + r) * HH + n0 + col] = acc[r] + bv;
    }
    __threadfence_block();
    __syncthreads();
    // ---- scores[b,t] = enc[b,t,:] . q[b,:] -------------------------------
    for (int i = tid; i < BB * TT; i += 256) {
      const int b = i >> 9, t = i & 511;
      const float* er = enc + ((size_t)b * TT + t) * HH;
      const float* qr = q_s + (size_t)b * HH;
      float s = 0.0f;
      for (int c = 0; c < HH; ++c) s += er[c] * qr[c];
      sc[i] = s;
    }
    __threadfence_block();
    __syncthreads();
    // ---- softmax per row (threads 0..127, one batch row each) ------------
    if (tid < BB) {
      float* row = sc + (size_t)tid * TT;
      float m = -1e30f;
      for (int t = 0; t < TT; ++t) m = fmaxf(m, row[t]);
      float sum = 0.0f;
      for (int t = 0; t < TT; ++t) { const float e_ = __expf(row[t] - m); row[t] = e_; sum += e_; }
      const float inv = 1.0f / sum;
      for (int t = 0; t < TT; ++t) row[t] *= inv;
    }
    __threadfence_block();
    __syncthreads();
    // ---- ctx[b,c] = sum_t w[b,t] * enc[b,t,c] ----------------------------
    for (int i = tid; i < BB * HH; i += 256) {
      const int b = i >> 8, c = i & 255;
      const float* row = sc + (size_t)b * TT;
      const float* er  = enc + (size_t)b * TT * HH + c;
      float s = 0.0f;
      for (int t = 0; t < TT; ++t) s += row[t] * er[(size_t)t * HH];
      ctxb[i] = s;
    }
    __threadfence_block();
    __syncthreads();
    // ---- logits = (h1 + ctx) @ out_W^T + out_b ; write to d_out ----------
    for (int i = tid; i < BB * 32; i += 256) {
      const int b = i >> 5, v = i & 31;
      if (v < VV) {
        float s = outb[v];
        const float* wr = outW + (size_t)v * HH;
        for (int c = 0; c < HH; ++c) s += (h1f[(size_t)b * HH + c] + ctxb[(size_t)b * HH + c]) * wr[c];
        out[((size_t)b * STEPS + step) * VV + v] = s;
      }
    }
    __threadfence_block();
    __syncthreads();
    // ---- argmax feedback -------------------------------------------------
    if (tid < BB) {
      const float* lr = out + ((size_t)tid * STEPS + step) * VV;
      int best = 0; float bv = lr[0];
      for (int v = 1; v < VV; ++v) if (lr[v] > bv) { bv = lr[v]; best = v; }
      tok[tid] = best;
    }
    __threadfence_block();
    __syncthreads();
  }
  // hidden = stack([h0, h1])
  float* hid = out + (size_t)BB * STEPS * VV;
  for (int i = tid; i < BB * HH; i += 256) {
    hid[i] = h0f[i];
    hid[BB * HH + i] = h1f[i];
  }
}

// ---------------------------------------------------------------------------
// Host launcher
// ---------------------------------------------------------------------------
extern "C" void kernel_launch(void* const* d_in, const int* in_sizes, int n_in,
                              void* d_out, int out_size, void* d_ws, size_t ws_size,
                              hipStream_t stream) {
  (void)in_sizes; (void)n_in; (void)out_size; (void)ws_size;

  const float* x     = (const float*)d_in[0];
  const int*   tgt   = (const int*)d_in[1];
  const float* eWih0 = (const float*)d_in[2];
  const float* eWhh0 = (const float*)d_in[3];
  const float* ebih0 = (const float*)d_in[4];
  const float* ebhh0 = (const float*)d_in[5];
  const float* eWih1 = (const float*)d_in[6];
  const float* eWhh1 = (const float*)d_in[7];
  const float* ebih1 = (const float*)d_in[8];
  const float* ebhh1 = (const float*)d_in[9];
  const float* e2dW  = (const float*)d_in[10];
  const float* e2db  = (const float*)d_in[11];
  const float* dWih0 = (const float*)d_in[12];
  const float* dWih1 = (const float*)d_in[13];
  const float* dWhh  = (const float*)d_in[14];
  const float* dbih  = (const float*)d_in[15];
  const float* dbhh  = (const float*)d_in[16];
  const float* qW    = (const float*)d_in[17];
  const float* qb    = (const float*)d_in[18];
  const float* outW  = (const float*)d_in[19];
  const float* outb  = (const float*)d_in[20];
  const float* emb   = (const float*)d_in[21];

  char* w = (char*)d_ws;
  size_t off = 0;
  auto alloc = [&](size_t bytes) -> void* {
    void* p = w + off;
    off = (off + bytes + 255) & ~(size_t)255;
    return p;
  };

  // f16 weight buffers
  _Float16* Whh0h  = (_Float16*)alloc((size_t)2 * GG * HH * 2);
  _Float16* Wih1h  = (_Float16*)alloc((size_t)2 * GG * 512 * 2);
  _Float16* Whh1h  = (_Float16*)alloc((size_t)2 * GG * HH * 2);
  _Float16* e2dh   = (_Float16*)alloc((size_t)HH * 512 * 2);
  _Float16* dWhhh  = (_Float16*)alloc((size_t)2 * GG * HH * 2);
  _Float16* dWih1h = (_Float16*)alloc((size_t)GG * HH * 2);
  _Float16* qWh    = (_Float16*)alloc((size_t)HH * HH * 2);
  // activations / scratch
  _Float16* y0h    = (_Float16*)alloc((size_t)BB * TT * 512 * 2);
  _Float16* y1h    = (_Float16*)alloc((size_t)BB * TT * 512 * 2);
  float*    encout = (float*)alloc((size_t)BB * TT * HH * 4);
  float*    gh_l0  = (float*)alloc((size_t)2 * BB * GG * 4);
  float*    gi_l1  = (float*)alloc((size_t)2 * BB * GG * 4);
  float*    gh_l1  = (float*)alloc((size_t)2 * BB * GG * 4);
  float*    hT0    = (float*)alloc((size_t)2 * BB * HH * 4);
  float*    hT1    = (float*)alloc((size_t)2 * BB * HH * 4);
  // decoder state/scratch
  _Float16* h0h  = (_Float16*)alloc((size_t)BB * HH * 2);
  _Float16* h1h  = (_Float16*)alloc((size_t)BB * HH * 2);
  float*    h0f  = (float*)alloc((size_t)BB * HH * 4);
  float*    h1f  = (float*)alloc((size_t)BB * HH * 4);
  float*    gi_s = (float*)alloc((size_t)BB * GG * 4);
  float*    gh_s = (float*)alloc((size_t)BB * GG * 4);
  float*    q_s  = (float*)alloc((size_t)BB * HH * 4);
  float*    ctxb = (float*)alloc((size_t)BB * HH * 4);
  float*    scb  = (float*)alloc((size_t)BB * TT * 4);

  auto cvt = [&](const float* s, _Float16* d, int n) {
    cvt_f32_to_f16<<<(n + 255) / 256, 256, 0, stream>>>(s, d, n);
  };
  cvt(eWhh0, Whh0h, 2 * GG * HH);
  cvt(eWih1, Wih1h, 2 * GG * 512);
  cvt(eWhh1, Whh1h, 2 * GG * HH);
  cvt(e2dW,  e2dh,  HH * 512);
  cvt(dWhh,  dWhhh, 2 * GG * HH);
  cvt(dWih1, dWih1h, GG * HH);
  cvt(qW,    qWh,   HH * HH);

  // Encoder layer 0 (both directions concurrently, persistent scan)
  enc_scan_l0<<<2, 1024, 0, stream>>>(x, eWih0, ebih0, Whh0h, ebhh0,
                                      y0h, hT0, gh_l0);
  // Encoder layer 1
  enc_scan_l1<<<2, 1024, 0, stream>>>(y0h, Wih1h, ebih1, Whh1h, ebhh1,
                                      y1h, hT1, gi_l1, gh_l1);
  // enc_out = y1 @ e2d_W^T + e2d_b   (65536 x 256 x 512)
  wmma_gemm_abt<<<dim3((BB * TT) / 16, HH / 64), 128, 0, stream>>>(
      y1h, e2dh, e2db, encout, BB * TT, HH, 512);
  // Decoder (persistent, 63 steps, greedy feedback)
  decoder_kernel<<<1, 256, 0, stream>>>(
      tgt, emb, dWih0, dWih1h, dWhhh, dbih, dbhh, qWh, qb, outW, outb,
      hT0, hT1, encout, h0h, h1h, h0f, h1f, gi_s, gh_s, q_s, ctxb, scb,
      (float*)d_out);
}
